// GAT_22488448762008
// MI455X (gfx1250) — matrix operand order repaired and checked by reading, not compiled, not run on hardware
//
#include <hip/hip_runtime.h>
#include <hip/hip_bf16.h>

#define FEAT 64
#define NGRAPH 64
#define OUTD 10
#define NEG_SLOPE_F 0.2f
#define BN_EPS_F 1e-5f

typedef __attribute__((ext_vector_type(16))) _Float16 v16h;
typedef __attribute__((ext_vector_type(8)))  _Float16 v8h;
typedef __attribute__((ext_vector_type(8)))  float    v8f;

// ---------------- small utility kernels ----------------

__global__ void gat_init_emax(unsigned* emax_key) {
    if (threadIdx.x == 0 && blockIdx.x == 0) *emax_key = 0x007FFFFFu; // ordkey(-inf)
}

__global__ void gat_convert_f16(const float* __restrict__ src,
                                _Float16* __restrict__ dst, int n) {
    int i = blockIdx.x * blockDim.x + threadIdx.x;
    if (i < n) dst[i] = (_Float16)src[i];
}

// Pre-shuffle W (64x64 row-major, K x N) into WMMA B-fragment layout:
// frag[kt][nt][lane][j]  with  col = nt*16 + (lane&15),  K = kt*32 + (j>>3)*16 + (lane>>4)*8 + (j&7)
__global__ void gat_build_wfrag(const float* __restrict__ W,
                                _Float16* __restrict__ frag) {
    int idx = blockIdx.x * blockDim.x + threadIdx.x;
    if (idx >= 2 * 4 * 32 * 16) return;
    int j    = idx & 15;
    int lane = (idx >> 4) & 31;
    int nt   = (idx >> 9) & 3;
    int kt   = (idx >> 11) & 1;
    int n = nt * 16 + (lane & 15);
    int k = kt * 32 + (j >> 3) * 16 + (lane >> 4) * 8 + (j & 7);
    frag[idx] = (_Float16)W[k * 64 + n];
}

// ---------------- WMMA GEMM: H[N,64] = A[N,64] (f16) x W[64,64] ----------------
// 256 threads = 8 waves; each wave computes a 16x64 stripe via 2 k-tiles x 4 n-tiles.
__global__ void gat_gemm_wmma(const _Float16* __restrict__ A,
                              const _Float16* __restrict__ Bfrag,
                              float* __restrict__ H, int nrows) {
    int wave = threadIdx.x >> 5;
    int lane = threadIdx.x & 31;
    int tile = blockIdx.x * 8 + wave;            // 16-row tile index (wave-uniform)
    if (tile * 16 + 16 > nrows) return;          // EXEC stays all-ones in active waves
    int row0 = tile * 16;

    int m  = lane & 15;
    int kh = lane >> 4;
    const _Float16* arow = A + (size_t)(row0 + m) * 64;

    v8f c[4] = {};
#pragma unroll
    for (int kt = 0; kt < 2; ++kt) {
        // A 16x32 f16 fragment: elems 0..7 = K kt*32+kh*8.. ; elems 8..15 = +16
        v8h lo = *(const v8h*)(arow + kt * 32 + kh * 8);
        v8h hi = *(const v8h*)(arow + kt * 32 + 16 + kh * 8);
        v16h a;
#pragma unroll
        for (int i = 0; i < 8; ++i) { a[i] = lo[i]; a[i + 8] = hi[i]; }
#pragma unroll
        for (int nt = 0; nt < 4; ++nt) {
            v16h b = *(const v16h*)(Bfrag + (((kt * 4 + nt) * 32 + lane) << 4));
            c[nt] = __builtin_amdgcn_wmma_f32_16x16x32_f16(
                false, a, false, b, (short)0, c[nt], false, false);
        }
    }
    // C/D layout: VGPR r -> row = r + (lane>=16 ? 8 : 0), col = lane&15
    int colb  = lane & 15;
    int rbase = (lane >> 4) * 8;
#pragma unroll
    for (int nt = 0; nt < 4; ++nt) {
#pragma unroll
        for (int r = 0; r < 8; ++r) {
            H[(size_t)(row0 + rbase + r) * 64 + nt * 16 + colb] = c[nt][r];
        }
    }
}

// ---------------- attention scores per node ----------------
__global__ void gat_scores(const float* __restrict__ H,
                           const float* __restrict__ a_s,
                           const float* __restrict__ a_d,
                           float* __restrict__ ssrc, float* __restrict__ sdst, int n) {
    int i = blockIdx.x * blockDim.x + threadIdx.x;
    if (i >= n) return;
    const float* h = H + (size_t)i * 64;
    float s0 = 0.f, s1 = 0.f;
#pragma unroll
    for (int k = 0; k < 64; ++k) { float v = h[k]; s0 += v * a_s[k]; s1 += v * a_d[k]; }
    ssrc[i] = s0; sdst[i] = s1;
}

// ---------------- edge softmax ----------------
__device__ __forceinline__ unsigned gat_ordkey(float x) {
    unsigned b = __float_as_uint(x);
    return (b & 0x80000000u) ? ~b : (b | 0x80000000u);
}

__global__ void gat_edge_max(const int* __restrict__ src, const int* __restrict__ dst,
                             const float* __restrict__ ssrc, const float* __restrict__ sdst,
                             unsigned* emax_key, int E) {
    __shared__ float smax[256];
    int e = blockIdx.x * blockDim.x + threadIdx.x;
    float v = -INFINITY;
    if (e < E) {
        float x = ssrc[src[e]] + sdst[dst[e]];
        v = (x >= 0.f) ? x : NEG_SLOPE_F * x;
    }
    smax[threadIdx.x] = v;
    __syncthreads();
    for (int s = 128; s > 0; s >>= 1) {
        if (threadIdx.x < s) smax[threadIdx.x] = fmaxf(smax[threadIdx.x], smax[threadIdx.x + s]);
        __syncthreads();
    }
    if (threadIdx.x == 0) atomicMax(emax_key, gat_ordkey(smax[0]));
}

__global__ void gat_edge_exp(const int* __restrict__ src, const int* __restrict__ dst,
                             const float* __restrict__ ssrc, const float* __restrict__ sdst,
                             const unsigned* emax_key,
                             float* __restrict__ wexp, float* __restrict__ denom, int E) {
    int e = blockIdx.x * blockDim.x + threadIdx.x;
    if (e >= E) return;
    unsigned key = *emax_key;
    unsigned bits = (key & 0x80000000u) ? (key ^ 0x80000000u) : ~key;
    float emax = __uint_as_float(bits);
    float x = ssrc[src[e]] + sdst[dst[e]];
    x = (x >= 0.f) ? x : NEG_SLOPE_F * x;
    float w = expf(x - emax);
    wexp[e] = w;
    atomicAdd(&denom[dst[e]], w);
}

// one (edge, feature) pair per thread; 4 edges per 256-thread block
__global__ void gat_aggregate(const int* __restrict__ src, const int* __restrict__ dst,
                              const float* __restrict__ wexp, const float* __restrict__ denom,
                              const float* __restrict__ H, float* __restrict__ agg, int E) {
    int t = threadIdx.x;
    int e = blockIdx.x * 4 + (t >> 6);
    if (e >= E) return;
    int f = t & 63;
    int s = src[e], d = dst[e];
    float alpha = wexp[e] / (denom[d] + 1e-12f);
    atomicAdd(&agg[(size_t)d * 64 + f], H[(size_t)s * 64 + f] * alpha);
}

// ---------------- batchnorm stats / apply+ELU+pool ----------------
__global__ void gat_bn_stats(const float* __restrict__ agg, float* __restrict__ bnstats, int n) {
    __shared__ float ssum[256], ssq[256];
    int f  = threadIdx.x & 63;
    int g4 = threadIdx.x >> 6;
    float sum = 0.f, sq = 0.f;
    for (int i = blockIdx.x * 4 + g4; i < n; i += gridDim.x * 4) {
        float v = agg[(size_t)i * 64 + f];
        sum += v; sq += v * v;
    }
    ssum[threadIdx.x] = sum; ssq[threadIdx.x] = sq;
    __syncthreads();
    if (g4 == 0) {
        sum = ssum[f] + ssum[f + 64] + ssum[f + 128] + ssum[f + 192];
        sq  = ssq[f]  + ssq[f + 64]  + ssq[f + 128]  + ssq[f + 192];
        atomicAdd(&bnstats[f], sum);
        atomicAdd(&bnstats[64 + f], sq);
    }
}

__global__ void gat_bn_elu_pool(float* __restrict__ feat, const float* __restrict__ bnstats,
                                const float* __restrict__ gamma, const float* __restrict__ beta,
                                const int* __restrict__ gid, float* __restrict__ pooled_l, int n) {
    int idx = blockIdx.x * blockDim.x + threadIdx.x;
    if (idx >= n * 64) return;
    int i = idx >> 6, f = idx & 63;
    float invN = 1.0f / (float)n;
    float mu  = bnstats[f] * invN;
    float var = bnstats[64 + f] * invN - mu * mu;
    float v = (feat[idx] - mu) * rsqrtf(var + BN_EPS_F) * gamma[f] + beta[f];
    v = (v > 0.f) ? v : (expf(v) - 1.f);   // ELU
    feat[idx] = v;
    atomicAdd(&pooled_l[gid[i] * 64 + f], v);
}

__global__ void gat_pool_input(const float* __restrict__ X, const int* __restrict__ gid,
                               float* __restrict__ pooled0, float* __restrict__ counts, int n) {
    int idx = blockIdx.x * blockDim.x + threadIdx.x;
    if (idx >= n * 64) return;
    int i = idx >> 6, f = idx & 63;
    atomicAdd(&pooled0[gid[i] * 64 + f], X[idx]);
    if (f == 0) atomicAdd(&counts[gid[i]], 1.0f);
}

// ---------------- final prediction heads ----------------
__global__ void gat_final(const float* __restrict__ pooled, const float* __restrict__ counts,
                          const float* w0, const float* w1, const float* w2,
                          const float* w3, const float* w4,
                          const float* b0, const float* b1, const float* b2,
                          const float* b3, const float* b4,
                          float* __restrict__ out) {
    int t = blockIdx.x * blockDim.x + threadIdx.x;
    if (t >= NGRAPH * OUTD) return;
    int g = t / OUTD, o = t % OUTD;
    float inv = 1.0f / fmaxf(counts[g], 1.0f);
    const float* wl[5] = {w0, w1, w2, w3, w4};
    const float* bl[5] = {b0, b1, b2, b3, b4};
    float acc = 0.f;
    for (int l = 0; l < 5; ++l) {
        const float* p = pooled + (l * NGRAPH + g) * 64;
        float s = 0.f;
        for (int k = 0; k < 64; ++k) s += p[k] * wl[l][k * OUTD + o];
        acc += s * inv + bl[l][o];
    }
    out[t] = acc;
}

// ---------------- host orchestration ----------------
extern "C" void kernel_launch(void* const* d_in, const int* in_sizes, int n_in,
                              void* d_out, int out_size, void* d_ws, size_t ws_size,
                              hipStream_t stream) {
    const int N = in_sizes[0] / FEAT;     // 50000
    const int E = in_sizes[1] / 2;        // 850000

    const float* X   = (const float*)d_in[0];
    const int* srcI  = (const int*)d_in[1];
    const int* dstI  = srcI + E;
    const int* gid   = (const int*)d_in[2];
    // W_list 3..6, a_src 7..10, a_dst 11..14, bn_gamma 15..18, bn_beta 19..22,
    // pred_W 23..27, pred_b 28..32

    char* ws = (char*)d_ws;
    size_t off = 0;
    auto take = [&](size_t bytes) -> void* {
        void* p = ws + off;
        off = (off + bytes + 255) & ~(size_t)255;
        return p;
    };
    float*     feat   = (float*)take((size_t)N * 64 * 4);   // layer input & agg target
    _Float16*  hf16   = (_Float16*)take((size_t)N * 64 * 2);
    float*     Hbuf   = (float*)take((size_t)N * 64 * 4);
    _Float16*  wfrag  = (_Float16*)take(4096 * 2);
    float*     ssrc   = (float*)take((size_t)N * 4);
    float*     sdst   = (float*)take((size_t)N * 4);
    float*     wexp   = (float*)take((size_t)E * 4);
    float*     denom  = (float*)take((size_t)N * 4);
    unsigned*  emaxk  = (unsigned*)take(256);
    float*     bnstats= (float*)take(128 * 4);
    float*     pooled = (float*)take(5 * NGRAPH * 64 * 4);
    float*     counts = (float*)take(NGRAPH * 4);

    const int NF = N * 64;

    hipMemsetAsync(pooled, 0, 5 * NGRAPH * 64 * 4, stream);
    hipMemsetAsync(counts, 0, NGRAPH * 4, stream);
    gat_pool_input<<<(NF + 255) / 256, 256, 0, stream>>>(X, gid, pooled, counts, N);

    const float* hsrc = X;
    for (int l = 0; l < 4; ++l) {
        gat_convert_f16<<<(NF + 255) / 256, 256, 0, stream>>>(hsrc, hf16, NF);
        hipMemsetAsync(feat, 0, (size_t)NF * 4, stream);
        hipMemsetAsync(denom, 0, (size_t)N * 4, stream);
        hipMemsetAsync(bnstats, 0, 128 * 4, stream);
        gat_init_emax<<<1, 32, 0, stream>>>(emaxk);

        gat_build_wfrag<<<16, 256, 0, stream>>>((const float*)d_in[3 + l], wfrag);

        int ntiles = N / 16;                             // N divisible by 16
        gat_gemm_wmma<<<(ntiles + 7) / 8, 256, 0, stream>>>(hf16, wfrag, Hbuf, N);

        gat_scores<<<(N + 255) / 256, 256, 0, stream>>>(
            Hbuf, (const float*)d_in[7 + l], (const float*)d_in[11 + l], ssrc, sdst, N);

        gat_edge_max<<<(E + 255) / 256, 256, 0, stream>>>(srcI, dstI, ssrc, sdst, emaxk, E);
        gat_edge_exp<<<(E + 255) / 256, 256, 0, stream>>>(srcI, dstI, ssrc, sdst, emaxk,
                                                          wexp, denom, E);
        gat_aggregate<<<(E + 3) / 4, 256, 0, stream>>>(srcI, dstI, wexp, denom, Hbuf, feat, E);

        gat_bn_stats<<<512, 256, 0, stream>>>(feat, bnstats, N);
        gat_bn_elu_pool<<<(NF + 255) / 256, 256, 0, stream>>>(
            feat, bnstats, (const float*)d_in[15 + l], (const float*)d_in[19 + l],
            gid, pooled + (l + 1) * NGRAPH * 64, N);

        hsrc = feat;
    }

    gat_final<<<(NGRAPH * OUTD + 255) / 256, 256, 0, stream>>>(
        pooled, counts,
        (const float*)d_in[23], (const float*)d_in[24], (const float*)d_in[25],
        (const float*)d_in[26], (const float*)d_in[27],
        (const float*)d_in[28], (const float*)d_in[29], (const float*)d_in[30],
        (const float*)d_in[31], (const float*)d_in[32],
        (float*)d_out);
}
